// SimpleGATConv_14353780704094
// MI455X (gfx1250) — compile-verified, matrix operand order checked by reading, not run on hardware
//
#include <hip/hip_runtime.h>

typedef __attribute__((ext_vector_type(16))) _Float16 v16h;
typedef __attribute__((ext_vector_type(8)))  float    v8f;

#define HEADS 4
#define OUTC 32
#define HC 128            // HEADS*OUTC == IN_CH == 128
#define NEG_SLOPE 0.2f
#define EPS_F 1e-9f

// ---- monotonic float<->uint encoding for atomicMax-based segment max ----
__device__ __forceinline__ unsigned enc_f(float f) {
  unsigned u = __float_as_uint(f);
  return (u & 0x80000000u) ? ~u : (u | 0x80000000u);
}
__device__ __forceinline__ float dec_f(unsigned u) {
  return (u & 0x80000000u) ? __uint_as_float(u & 0x7FFFFFFFu)
                           : __uint_as_float(~u);
}

__device__ __forceinline__ float leaky(float v) {
  return v >= 0.0f ? v : NEG_SLOPE * v;
}

// =====================  1) weight f32 -> f16  =====================
__global__ void gat_cast_w(const float* __restrict__ w, _Float16* __restrict__ w16) {
  int i = blockIdx.x * blockDim.x + threadIdx.x;
  if (i < HC * HC) w16[i] = (_Float16)w[i];
}

// =====================  2) pre-build WMMA B fragments  =====================
// B-frag for col-tile t (n0=t*16), k-step s (k0=s*32):
// lane = hiHalf*16 + n  holds W[k0 + kmap(j,hi)][t*16 + n], j=0..15
// kmap(j,hi) = (j<8) ? hi*8+j : 16+hi*8+(j-8)   (mirrors the 16-bit A layout)
__global__ void gat_build_wfrag(const _Float16* __restrict__ w16,
                                _Float16* __restrict__ wfrag) {
  int i = blockIdx.x * blockDim.x + threadIdx.x; // 0..1023
  if (i >= 8 * 4 * 32) return;
  int lane = i & 31;
  int s    = (i >> 5) & 3;
  int t    = i >> 7;
  int hi   = lane >> 4;
  int n    = t * 16 + (lane & 15);
  _Float16* o = wfrag + (size_t)i * 16;
#pragma unroll
  for (int j = 0; j < 16; ++j) {
    int k = s * 32 + ((j < 8) ? (hi * 8 + j) : (16 + hi * 8 + (j - 8)));
    o[j] = w16[k * HC + n];
  }
}

// =====================  3) h = x @ W via v_wmma_f32_16x16x32_f16  ===========
// block = 256 threads = 8 waves; wave w owns output col-tile n0 = w*16.
// block row-tile m0 = blockIdx.x*16.  K=128 = 4 WMMA steps of 32.
__global__ void gat_gemm_wmma(const float* __restrict__ x,
                              const _Float16* __restrict__ wfrag,
                              float* __restrict__ h, int N) {
  int wave = threadIdx.x >> 5;
  int lane = threadIdx.x & 31;
  int hi   = lane >> 4;
  int lm   = lane & 15;
  int m0   = blockIdx.x * 16;
  int n0   = wave * 16;

  int row = m0 + lm;
  if (row >= N) row = N - 1;           // clamp (N is a multiple of 16 anyway)
  const float* xr = x + (size_t)row * HC;

  v8f c = {};
#pragma unroll
  for (int s = 0; s < 4; ++s) {
    const float4 p0 = *(const float4*)(xr + s * 32 + hi * 8);
    const float4 p1 = *(const float4*)(xr + s * 32 + hi * 8 + 4);
    const float4 p2 = *(const float4*)(xr + s * 32 + 16 + hi * 8);
    const float4 p3 = *(const float4*)(xr + s * 32 + 16 + hi * 8 + 4);
    v16h a;
    a[0]=(_Float16)p0.x;  a[1]=(_Float16)p0.y;  a[2]=(_Float16)p0.z;  a[3]=(_Float16)p0.w;
    a[4]=(_Float16)p1.x;  a[5]=(_Float16)p1.y;  a[6]=(_Float16)p1.z;  a[7]=(_Float16)p1.w;
    a[8]=(_Float16)p2.x;  a[9]=(_Float16)p2.y;  a[10]=(_Float16)p2.z; a[11]=(_Float16)p2.w;
    a[12]=(_Float16)p3.x; a[13]=(_Float16)p3.y; a[14]=(_Float16)p3.z; a[15]=(_Float16)p3.w;

    v16h b = *(const v16h*)(wfrag + ((size_t)(wave * 4 + s) * 32 + lane) * 16);

    c = __builtin_amdgcn_wmma_f32_16x16x32_f16(
            /*neg_a=*/false, a, /*neg_b=*/false, b,
            /*c_mod=*/(short)0, c, /*reuse_a=*/false, /*reuse_b=*/false);
  }

  // D layout: element i -> D[M = hi*8 + i][N = lm]
#pragma unroll
  for (int i = 0; i < 8; ++i) {
    int r = m0 + hi * 8 + i;
    if (r < N) h[(size_t)r * HC + n0 + lm] = c[i];
  }
}

// =====================  4) per-node attention logits  =====================
// one wave per node; lane owns 4 channels; heads = lane/8 groups; shfl-reduce.
__global__ void gat_logits(const float* __restrict__ h,
                           const float* __restrict__ att_src,
                           const float* __restrict__ att_dst,
                           float* __restrict__ a_s, float* __restrict__ a_d, int N) {
  int node = blockIdx.x * 8 + (threadIdx.x >> 5);
  int lane = threadIdx.x & 31;
  if (node >= N) return;
  float4 hv = *(const float4*)(h + (size_t)node * HC + lane * 4);
  float4 as = *(const float4*)(att_src + lane * 4);   // lane*4 == head*32 + (lane%8)*4
  float4 ad = *(const float4*)(att_dst + lane * 4);
  float ps = hv.x * as.x + hv.y * as.y + hv.z * as.z + hv.w * as.w;
  float pd = hv.x * ad.x + hv.y * ad.y + hv.z * ad.z + hv.w * ad.w;
#pragma unroll
  for (int off = 1; off < 8; off <<= 1) {
    ps += __shfl_xor(ps, off, 32);
    pd += __shfl_xor(pd, off, 32);
  }
  if ((lane & 7) == 0) {
    int head = lane >> 3;
    a_s[node * HEADS + head] = ps;
    a_d[node * HEADS + head] = pd;
  }
}

// =====================  5) init accumulators  =====================
__global__ void gat_init(float* __restrict__ out, unsigned* __restrict__ maxbuf,
                         float* __restrict__ denom, int nOut, int nNH) {
  int i = blockIdx.x * blockDim.x + threadIdx.x;
  if (i < nOut) out[i] = 0.0f;
  if (i < nNH) { maxbuf[i] = 0u; denom[i] = 0.0f; }   // 0 == enc(-inf)-ish floor
}

// =====================  6) pass 1: segment max  =====================
__global__ void gat_edge_max(const long long* __restrict__ ei,
                             const float* __restrict__ a_s,
                             const float* __restrict__ a_d,
                             unsigned* __restrict__ maxbuf, int E, int T) {
  int e = blockIdx.x * blockDim.x + threadIdx.x;
  if (e >= T) return;
  long long s, d;
  if (e < E) { s = ei[e]; d = ei[E + e]; } else { s = d = (long long)(e - E); }
  float4 as = *(const float4*)(a_s + s * HEADS);
  float4 ad = *(const float4*)(a_d + d * HEADS);
  unsigned* mb = maxbuf + d * HEADS;
  atomicMax(mb + 0, enc_f(leaky(as.x + ad.x)));
  atomicMax(mb + 1, enc_f(leaky(as.y + ad.y)));
  atomicMax(mb + 2, enc_f(leaky(as.z + ad.z)));
  atomicMax(mb + 3, enc_f(leaky(as.w + ad.w)));
}

// =====================  7) pass 2: segment sum of exp  =====================
__global__ void gat_edge_sum(const long long* __restrict__ ei,
                             const float* __restrict__ a_s,
                             const float* __restrict__ a_d,
                             const unsigned* __restrict__ maxbuf,
                             float* __restrict__ denom, int E, int T) {
  int e = blockIdx.x * blockDim.x + threadIdx.x;
  if (e >= T) return;
  long long s, d;
  if (e < E) { s = ei[e]; d = ei[E + e]; } else { s = d = (long long)(e - E); }
  float4 as = *(const float4*)(a_s + s * HEADS);
  float4 ad = *(const float4*)(a_d + d * HEADS);
  const unsigned* mb = maxbuf + d * HEADS;
  float* dn = denom + d * HEADS;
  atomicAdd(dn + 0, __expf(leaky(as.x + ad.x) - dec_f(mb[0])));
  atomicAdd(dn + 1, __expf(leaky(as.y + ad.y) - dec_f(mb[1])));
  atomicAdd(dn + 2, __expf(leaky(as.z + ad.z) - dec_f(mb[2])));
  atomicAdd(dn + 3, __expf(leaky(as.w + ad.w) - dec_f(mb[3])));
}

// =====================  8) pass 3: weighted scatter-add  =====================
// one wave per edge; lane owns 4 channels (float4 gather of h[src], 4 atomics).
__global__ void gat_edge_agg(const long long* __restrict__ ei,
                             const float* __restrict__ a_s,
                             const float* __restrict__ a_d,
                             const unsigned* __restrict__ maxbuf,
                             const float* __restrict__ denom,
                             const float* __restrict__ h,
                             float* __restrict__ out, int E, int T) {
  int e = blockIdx.x * 8 + (threadIdx.x >> 5);
  int lane = threadIdx.x & 31;
  if (e >= T) return;
  long long s, d;
  if (e < E) { s = ei[e]; d = ei[E + e]; } else { s = d = (long long)(e - E); }
  int head = lane >> 3;
  float v  = leaky(a_s[s * HEADS + head] + a_d[d * HEADS + head]);
  float mx = dec_f(maxbuf[d * HEADS + head]);
  float alpha = __expf(v - mx) / (denom[d * HEADS + head] + EPS_F);
  float4 hv = *(const float4*)(h + s * HC + lane * 4);
  float* o = out + d * HC + lane * 4;
  atomicAdd(o + 0, alpha * hv.x);
  atomicAdd(o + 1, alpha * hv.y);
  atomicAdd(o + 2, alpha * hv.z);
  atomicAdd(o + 3, alpha * hv.w);
}

// =====================  9) bias  =====================
__global__ void gat_bias(float* __restrict__ out, const float* __restrict__ bias, int n) {
  int i = blockIdx.x * blockDim.x + threadIdx.x;
  if (i < n) out[i] += bias[i & (HC - 1)];
}

extern "C" void kernel_launch(void* const* d_in, const int* in_sizes, int n_in,
                              void* d_out, int out_size, void* d_ws, size_t ws_size,
                              hipStream_t stream) {
  const float*     x       = (const float*)d_in[0];
  const long long* ei      = (const long long*)d_in[1];   // int64 edge_index [2,E]
  const float*     weight  = (const float*)d_in[2];
  const float*     att_src = (const float*)d_in[3];
  const float*     att_dst = (const float*)d_in[4];
  const float*     bias    = (const float*)d_in[5];
  float*           out     = (float*)d_out;

  const int N = in_sizes[0] / HC;        // 50000
  const int E = in_sizes[1] / 2;         // 1600000
  const int T = E + N;                   // edges + self loops

  // workspace carve-out (hipMalloc base is 256B aligned)
  char* ws = (char*)d_ws;
  size_t off = 0;
  auto take = [&](size_t bytes) { char* p = ws + off; off = (off + bytes + 255) & ~(size_t)255; return p; };
  _Float16* w16    = (_Float16*)take((size_t)HC * HC * sizeof(_Float16));
  _Float16* wfrag  = (_Float16*)take((size_t)8 * 4 * 32 * 16 * sizeof(_Float16));
  float*    h      = (float*)   take((size_t)N * HC * sizeof(float));
  float*    a_s    = (float*)   take((size_t)N * HEADS * sizeof(float));
  float*    a_d    = (float*)   take((size_t)N * HEADS * sizeof(float));
  unsigned* maxbuf = (unsigned*)take((size_t)N * HEADS * sizeof(unsigned));
  float*    denom  = (float*)   take((size_t)N * HEADS * sizeof(float));
  (void)ws_size;

  const int B = 256;
  gat_cast_w     <<<(HC * HC + B - 1) / B, B, 0, stream>>>(weight, w16);
  gat_build_wfrag<<<(8 * 4 * 32 + B - 1) / B, B, 0, stream>>>(w16, wfrag);
  gat_gemm_wmma  <<<(N + 15) / 16, B, 0, stream>>>(x, wfrag, h, N);
  gat_logits     <<<(N + 7) / 8, B, 0, stream>>>(h, att_src, att_dst, a_s, a_d, N);

  const int nOut = N * HC, nNH = N * HEADS;
  gat_init       <<<(nOut + B - 1) / B, B, 0, stream>>>(out, maxbuf, denom, nOut, nNH);
  gat_edge_max   <<<(T + B - 1) / B, B, 0, stream>>>(ei, a_s, a_d, maxbuf, E, T);
  gat_edge_sum   <<<(T + B - 1) / B, B, 0, stream>>>(ei, a_s, a_d, maxbuf, denom, E, T);
  gat_edge_agg   <<<(T + 7) / 8, B, 0, stream>>>(ei, a_s, a_d, maxbuf, denom, h, out, E, T);
  gat_bias       <<<(nOut + B - 1) / B, B, 0, stream>>>(out, bias, nOut);
}
